// HawkesDecoder_63711544868921
// MI455X (gfx1250) — matrix operand order repaired and checked by reading, not compiled, not run on hardware
//
#include <hip/hip_runtime.h>
#include <hip/hip_bf16.h>
#include <stdint.h>

// ---------------------------------------------------------------------------
// Hawkes continuous-time LSTM decoder for MI455X (gfx1250, wave32, WMMA).
// B=64, T=512, H=512, E=128, C=1000, G=7H=3584, K_total=E+H=640.
// Recurrent GEMM per step: (64 x 640) @ (640 x 3584) in bf16 -> f32 via
// v_wmma_f32_16x16x32_bf16. Weights (4.6 MB bf16) are L2-resident.
// ---------------------------------------------------------------------------

#define BB   64
#define TT   512
#define HH   512
#define EE   128
#define GG   3584          // 7*H
#define KK   640           // E+H
#define OUT_STRIDE 3072    // 6*H
#define OUT_T (TT + 1)

typedef __attribute__((ext_vector_type(16))) __bf16 v16bf;
typedef __attribute__((ext_vector_type(8)))  __bf16 v8bf;
typedef __attribute__((ext_vector_type(8)))  float  v8f;

// ----------------------------- helpers -------------------------------------

__device__ __forceinline__ unsigned short f2bf(float f) {
    unsigned u = __builtin_bit_cast(unsigned, f);
    unsigned r = u + 0x7FFFu + ((u >> 16) & 1u);   // round-to-nearest-even
    return (unsigned short)(r >> 16);
}

__device__ __forceinline__ float sigmoid_f(float x) {
    return 1.0f / (1.0f + __expf(-x));
}

__device__ __forceinline__ float softplus_f(float x) {
    // numerically stable log1p(exp(x))
    return fmaxf(x, 0.0f) + log1pf(__expf(-fabsf(x)));
}

__device__ __forceinline__ v16bf load_a_frag(const unsigned short* p0,
                                             const unsigned short* p1) {
    // A 16x32 bf16 fragment: lane half 0 -> K{0..15}, half 1 -> K{16..31}
    v8bf lo = *(const v8bf*)p0;   // 16B aligned
    v8bf hi = *(const v8bf*)p1;
    v16bf a;
#pragma unroll
    for (int i = 0; i < 8; ++i) { a[i] = lo[i]; a[i + 8] = hi[i]; }
    return a;
}

__device__ __forceinline__ v16bf load_b_frag(const unsigned short* p) {
    // B 32x16 bf16 fragment: lane = N, half selects K 0..15 / 16..31,
    // W stored N-major so the 16 K-values are 32 contiguous bytes.
    return *(const v16bf*)p;      // 32B aligned
}

// ----------------------- one-time conversion kernels ------------------------

// W_cell (KK x GG) f32 row-major  ->  Wt (GG x KK) bf16 (transposed, N-major)
__global__ void convert_wt_kernel(const float* __restrict__ W,
                                  unsigned short* __restrict__ Wt) {
    int idx = blockIdx.x * blockDim.x + threadIdx.x;
    if (idx >= GG * KK) return;
    int n = idx / KK;
    int k = idx - n * KK;
    Wt[idx] = f2bf(W[(size_t)k * GG + n]);
}

// xb[b,t,e] = bf16(emb_table[marks[b,t], e])
__global__ void gather_emb_kernel(const int* __restrict__ marks,
                                  const float* __restrict__ emb,
                                  unsigned short* __restrict__ xb) {
    int idx = blockIdx.x * blockDim.x + threadIdx.x;
    if (idx >= BB * TT * EE) return;
    int e  = idx & (EE - 1);
    int bt = idx >> 7;
    int m  = marks[bt];
    xb[idx] = f2bf(emb[(size_t)m * EE + e]);
}

// ----------------------------- init state -----------------------------------

__global__ void init_state_kernel(const float* __restrict__ init_hidden,
                                  float* __restrict__ c_d,
                                  float* __restrict__ c_bar,
                                  unsigned short* __restrict__ h_bf,
                                  float* __restrict__ out) {
    int idx = blockIdx.x * blockDim.x + threadIdx.x;   // 0 .. B*H-1
    if (idx >= BB * HH) return;
    int b = idx >> 9;            // /512
    int j = idx & (HH - 1);

    float hd = tanhf(init_hidden[j]);
    float cd = tanhf(init_hidden[HH + j]);
    float cb = tanhf(init_hidden[2 * HH + j]);
    float c0 = tanhf(init_hidden[3 * HH + j]);
    float d0 = softplus_f(init_hidden[4 * HH + j]);
    float o0 = sigmoid_f(init_hidden[5 * HH + j]);

    c_d[idx]  = cd;
    c_bar[idx] = cb;
    h_bf[idx] = f2bf(hd);

    size_t ob = (size_t)b * OUT_T * OUT_STRIDE;        // t = 0
    out[ob + j]            = hd;   // h_d
    out[ob + HH + j]       = o0;   // o
    out[ob + 2 * HH + j]   = cb;   // c_bar
    out[ob + 3 * HH + j]   = c0;   // c
    out[ob + 4 * HH + j]   = d0;   // delta
    out[ob + 5 * HH + j]   = cd;   // c_d
}

// --------------------------- recurrent GEMM ---------------------------------
// g = [x_t , h_{t-1}] @ W_cell + b_cell     (64 x 640) @ (640 x 3584)
// Block: 128 threads = 4 waves; block tile 64x64; wave tile 16x64 (4 WMMA acc).
__global__ void __launch_bounds__(128)
gemm_step_kernel(const unsigned short* __restrict__ xb,   // (B,T,E) bf16
                 const unsigned short* __restrict__ h_bf, // (B,H)  bf16
                 const unsigned short* __restrict__ Wt,   // (G,K)  bf16 N-major
                 const float* __restrict__ bias,          // (G)
                 float* __restrict__ g,                   // (B,G)
                 int t) {
    const int lane = threadIdx.x & 31;
    const int wave = threadIdx.x >> 5;         // 0..3 -> rows 16*wave..+15
    const int r    = lane & 15;                // A: row in tile / B: N in tile
    const int half = lane >> 4;                // K-half selector
    const int nBlock = blockIdx.x * 64;

    v8f acc0 = {}, acc1 = {}, acc2 = {}, acc3 = {};

    const int rowM = wave * 16 + r;            // batch row 0..63
    const unsigned short* aEmb = xb + ((size_t)rowM * TT + t) * EE;
    const unsigned short* aH   = h_bf + (size_t)rowM * HH;

    // --- K over embedding part (k = 0..127) ---
#pragma unroll
    for (int k0 = 0; k0 < EE; k0 += 32) {
        v16bf a = load_a_frag(aEmb + k0 + half * 16,
                              aEmb + k0 + 8 + half * 16);
#pragma unroll
        for (int j = 0; j < 4; ++j) {
            const unsigned short* bp =
                Wt + (size_t)(nBlock + j * 16 + r) * KK + k0 + half * 16;
            __builtin_prefetch(bp + 32, 0, 1);
            v16bf bfrag = load_b_frag(bp);
            v8f* accp = (j == 0) ? &acc0 : (j == 1) ? &acc1 : (j == 2) ? &acc2 : &acc3;
            *accp = __builtin_amdgcn_wmma_f32_16x16x32_bf16(
                false, a, false, bfrag, (short)0, *accp, false, false);
        }
    }

    // --- K over hidden part (k = 128..639) ---
    for (int k0 = 0; k0 < HH; k0 += 32) {
        v16bf a = load_a_frag(aH + k0 + half * 16,
                              aH + k0 + 8 + half * 16);
#pragma unroll
        for (int j = 0; j < 4; ++j) {
            const unsigned short* bp =
                Wt + (size_t)(nBlock + j * 16 + r) * KK + EE + k0 + half * 16;
            __builtin_prefetch(bp + 32, 0, 1);
            v16bf bfrag = load_b_frag(bp);
            v8f* accp = (j == 0) ? &acc0 : (j == 1) ? &acc1 : (j == 2) ? &acc2 : &acc3;
            *accp = __builtin_amdgcn_wmma_f32_16x16x32_bf16(
                false, a, false, bfrag, (short)0, *accp, false, false);
        }
    }

    // --- store D + bias: lane holds N = r, VGPR v holds M = v + 8*half ---
#pragma unroll
    for (int j = 0; j < 4; ++j) {
        const v8f acc = (j == 0) ? acc0 : (j == 1) ? acc1 : (j == 2) ? acc2 : acc3;
        const int col = nBlock + j * 16 + r;
        const float bv = bias[col];
#pragma unroll
        for (int v = 0; v < 8; ++v) {
            const int row = wave * 16 + half * 8 + v;
            g[(size_t)row * GG + col] = acc[v] + bv;
        }
    }
}

// ------------------------- elementwise state update --------------------------

__global__ void step_update_kernel(const float* __restrict__ g,    // (B,G)
                                   const float* __restrict__ ts,   // (B,T)
                                   float* __restrict__ c_d,
                                   float* __restrict__ c_bar,
                                   unsigned short* __restrict__ h_bf,
                                   float* __restrict__ out,
                                   int t) {
    int idx = blockIdx.x * blockDim.x + threadIdx.x;   // 0 .. B*H-1
    if (idx >= BB * HH) return;
    int b = idx >> 9;
    int j = idx & (HH - 1);

    const float* gr = g + (size_t)b * GG;
    float gi  = sigmoid_f(gr[j]);
    float gf  = sigmoid_f(gr[HH + j]);
    float gz  = tanhf    (gr[2 * HH + j]);
    float go  = sigmoid_f(gr[3 * HH + j]);
    float gib = sigmoid_f(gr[4 * HH + j]);
    float gfb = sigmoid_f(gr[5 * HH + j]);
    float gd  = softplus_f(gr[6 * HH + j]);

    float tcur = ts[(size_t)b * TT + t];
    float dur  = (t > 0) ? (tcur - ts[(size_t)b * TT + t - 1]) : tcur;

    float cd = c_d[idx];
    float cb = c_bar[idx];

    float c   = gf * cd + gi * gz;
    float cbt = gfb * cb + gib * gz;
    float cdt = cbt + (c - cbt) * __expf(-gd * dur);
    float hdt = go * tanhf(cdt);

    c_d[idx]   = cdt;
    c_bar[idx] = cbt;
    h_bf[idx]  = f2bf(hdt);

    size_t ob = ((size_t)b * OUT_T + (t + 1)) * OUT_STRIDE;
    out[ob + j]          = hdt;   // h_d_t
    out[ob + HH + j]     = go;    // o
    out[ob + 2 * HH + j] = cbt;   // c_bar
    out[ob + 3 * HH + j] = c;     // c
    out[ob + 4 * HH + j] = gd;    // delta
    out[ob + 5 * HH + j] = cdt;   // c_d_t
}

// ------------------------------- launcher ------------------------------------

extern "C" void kernel_launch(void* const* d_in, const int* in_sizes, int n_in,
                              void* d_out, int out_size, void* d_ws, size_t ws_size,
                              hipStream_t stream) {
    const int*   marks = (const int*)  d_in[0];   // (B,T) int32
    const float* ts    = (const float*)d_in[1];   // (B,T)
    const float* emb   = (const float*)d_in[2];   // (C,E)
    const float* W     = (const float*)d_in[3];   // (E+H, 7H)
    const float* bias  = (const float*)d_in[4];   // (7H)
    const float* ih    = (const float*)d_in[5];   // (1, 6H)
    float* out = (float*)d_out;                   // (B, T+1, 6H)

    // Workspace layout (all offsets 256B-aligned)
    char* ws = (char*)d_ws;
    unsigned short* Wt   = (unsigned short*)(ws);                     // 4,587,520 B
    unsigned short* xb   = (unsigned short*)(ws + 4587520);           // 8,388,608 B
    unsigned short* h_bf = (unsigned short*)(ws + 12976128);          //    65,536 B
    float*          c_d  = (float*)         (ws + 13041664);          //   131,072 B
    float*          c_br = (float*)         (ws + 13172736);          //   131,072 B
    float*          g    = (float*)         (ws + 13303808);          //   917,504 B
    // total: 14,221,312 B

    // 1) weights -> bf16 transposed (L2-resident for the whole scan)
    convert_wt_kernel<<<(GG * KK) / 256, 256, 0, stream>>>(W, Wt);
    // 2) embedding gather -> bf16
    gather_emb_kernel<<<(BB * TT * EE) / 256, 256, 0, stream>>>(marks, emb, xb);
    // 3) state 0
    init_state_kernel<<<(BB * HH) / 256, 256, 0, stream>>>(ih, c_d, c_br, h_bf, out);

    // 4) sequential scan: WMMA GEMM + elementwise decay update per step
    for (int t = 0; t < TT; ++t) {
        gemm_step_kernel<<<GG / 64, 128, 0, stream>>>(xb, h_bf, Wt, bias, g, t);
        step_update_kernel<<<(BB * HH) / 256, 256, 0, stream>>>(g, ts, c_d, c_br,
                                                                h_bf, out, t);
    }
}
